// ASMKPooling_46016279609384
// MI455X (gfx1250) — matrix operand order repaired and checked by reading, not compiled, not run on hardware
//
#include <hip/hip_runtime.h>
#include <hip/hip_bf16.h>
#include <math.h>

// Problem constants (from reference): B=16, N=4096, D=128, K=1024
#define BB   16
#define NN   4096
#define DD   128
#define KK   1024
#define MTOT (BB * NN)   // 65536 points

typedef __attribute__((ext_vector_type(16))) __bf16 v16bf;
typedef __attribute__((ext_vector_type(8)))  float  v8f;

#define WMMA_BF16(a, b, c) \
  __builtin_amdgcn_wmma_f32_16x16x32_bf16(false, (a), false, (b), (short)0, (c), false, false)

#define WAIT_ASYNC() asm volatile("s_wait_asynccnt 0" ::: "memory")

// ---------------------------------------------------------------------------
// Kernel 1: split centroids into bf16 hi/lo and compute ||c||^2 per centroid.
// One block (128 threads) per centroid row.
// ---------------------------------------------------------------------------
__global__ __launch_bounds__(128) void prep_centroids(
    const float* __restrict__ cen, __bf16* __restrict__ ch,
    __bf16* __restrict__ cl, float* __restrict__ cnorm2) {
  int k = blockIdx.x;
  int d = threadIdx.x;
  float v = cen[k * DD + d];
  __bf16 h = (__bf16)v;
  float hf = (float)h;
  ch[k * DD + d] = h;
  cl[k * DD + d] = (__bf16)(v - hf);
  float s = v * v;
  #pragma unroll
  for (int m = 16; m >= 1; m >>= 1) s += __shfl_xor(s, m, 32);
  __shared__ float rs[4];
  int lane = threadIdx.x & 31, wv = threadIdx.x >> 5;
  if (lane == 0) rs[wv] = s;
  __syncthreads();
  if (threadIdx.x == 0) cnorm2[k] = rs[0] + rs[1] + rs[2] + rs[3];
}

// ---------------------------------------------------------------------------
// Kernel 2: main WMMA kernel. Per block: 128 points x all 1024 centroids.
// Centroid tiles (16 centroids x 128 K, hi+lo bf16 = 8 KB) are staged into a
// double-buffered LDS region by the whole block using CDNA5 async-to-LDS
// copies (ASYNCcnt), overlapped with the 12 WMMAs of the current tile.
// Per wave: 16 points; A fragments (hi/lo bf16 split of f32 x) live in
// registers for the whole sweep. dist^2 = ||x||^2 + ||c||^2 - 2 x.c; we
// track min of (||c||^2 - 2 x.c) and fold ||x||^2 + sqrt in at the end.
// ---------------------------------------------------------------------------
__global__ __launch_bounds__(256) void asmk_main(
    const float* __restrict__ x, const __bf16* __restrict__ ch,
    const __bf16* __restrict__ cl, const float* __restrict__ cnorm2,
    float* __restrict__ mind, int* __restrict__ nearest) {
  // [buf][hi/lo][centroid-in-tile][k] ; 2*2*16*128*2B = 16 KB
  __shared__ __align__(32) __bf16 cbuf[2][2][16][DD];
  __shared__ float lds_xn[8 * 16];  // ||x||^2 per row, per wave

  int t      = threadIdx.x;
  int lane   = t & 31;
  int wave   = t >> 5;
  int m0     = (blockIdx.x * 8 + wave) * 16;  // base point index of this wave
  int row    = lane & 15;                     // A row / B column / C column
  int halfId = lane >> 4;                     // 0: lanes 0-15, 1: lanes 16-31

  // ---- Load A fragments (16x32 bf16 layout): lane<16 holds K c*32+{0..15};
  // lane>=16 holds K c*32+{16..31} of row `row`.
  const float* xrow = x + (size_t)(m0 + row) * DD;
  int koff = halfId ? 16 : 0;
  v16bf ah[4], al[4];
  float partial = 0.f;
  #pragma unroll
  for (int c = 0; c < 4; ++c) {
    int base = c * 32 + koff;
    float f[16];
    #pragma unroll
    for (int i = 0; i < 8; ++i) f[i] = xrow[base + i];
    #pragma unroll
    for (int i = 0; i < 8; ++i) f[8 + i] = xrow[base + 8 + i];
    #pragma unroll
    for (int i = 0; i < 16; ++i) {
      partial += f[i] * f[i];
      __bf16 h = (__bf16)f[i];
      ah[c][i] = h;
      al[c][i] = (__bf16)(f[i] - (float)h);
    }
  }
  // Row norm: each half-wave covered half the row's elements.
  float xn = partial + __shfl_xor(partial, 16, 32);
  if (halfId == 0) lds_xn[wave * 16 + row] = xn;

  // ---- Async staging machinery. LDS byte offset = low 32 bits of flat addr
  // async copy is GVS-mode: saddr(uniform
  // tile base) + per-thread 32-bit byte offset.
  unsigned cbase = (unsigned)(uintptr_t)(&cbuf[0][0][0][0]);
  unsigned voff  = (unsigned)t * 16u;  // this thread's 16B chunk of the 4KB tile

  #define STAGE_TILE(NT1, PBUF)                                               \
    do {                                                                      \
      uint64_t sh = (uint64_t)(uintptr_t)(ch + (size_t)(NT1) * 16 * DD);      \
      uint64_t sl = (uint64_t)(uintptr_t)(cl + (size_t)(NT1) * 16 * DD);      \
      unsigned lh = cbase + (unsigned)(PBUF) * 8192u + voff;                  \
      unsigned ll = lh + 4096u;                                               \
      asm volatile("global_load_async_to_lds_b128 %0, %1, %2\n\t"             \
                   "global_load_async_to_lds_b128 %3, %4, %5"                 \
                   :: "v"(lh), "v"(voff), "s"(sh),                            \
                      "v"(ll), "v"(voff), "s"(sl)                             \
                   : "memory");                                               \
    } while (0)

  // Prologue: stage tile 0 into buffer 0.
  STAGE_TILE(0, 0);
  WAIT_ASYNC();
  __syncthreads();

  float minv[8];
  int   mini[8];
  #pragma unroll
  for (int v = 0; v < 8; ++v) { minv[v] = 3.4e38f; mini[v] = 0; }

  // ---- Sweep all 64 column tiles of 16 centroids (double-buffered).
  for (int nt = 0; nt < KK / 16; ++nt) {
    int p = nt & 1;
    if (nt + 1 < KK / 16) STAGE_TILE(nt + 1, p ^ 1);  // overlap copy w/ compute

    // B 32x16 bf16 layout: lanes 0-15 hold K c*32+{0..15} of their column,
    // lanes 16-31 hold K c*32+{16..31}: one contiguous 32B LDS read each.
    const __bf16* bufh = &cbuf[p][0][0][0];
    const __bf16* bufl = &cbuf[p][1][0][0];
    v16bf bh[4], bl[4];
    #pragma unroll
    for (int c = 0; c < 4; ++c) {
      int ks = c * 32 + (halfId ? 16 : 0);
      bh[c] = *(const v16bf*)(bufh + row * DD + ks);
      bl[c] = *(const v16bf*)(bufl + row * DD + ks);
    }
    v8f acc = {0.f, 0.f, 0.f, 0.f, 0.f, 0.f, 0.f, 0.f};
    #pragma unroll
    for (int c = 0; c < 4; ++c) {
      // Split-product: hi*hi + lo*hi + hi*lo (lo*lo ~ 2^-18, dropped)
      acc = WMMA_BF16(ah[c], bh[c], acc);
      acc = WMMA_BF16(al[c], bh[c], acc);
      acc = WMMA_BF16(ah[c], bl[c], acc);
    }

    // C/D layout: VGPR v holds row (v + 8*halfId), column = ncol for this lane.
    int ncol = nt * 16 + row;
    float cn = cnorm2[ncol];
    #pragma unroll
    for (int v = 0; v < 8; ++v) {
      float s = cn - 2.0f * acc[v];
      if (s < minv[v]) { minv[v] = s; mini[v] = ncol; }  // strict < => earliest col per lane
    }

    WAIT_ASYNC();     // tile nt+1 resident before anyone reads it next iter
    __syncthreads();  // and nobody still reads buffer p^1 when it's recycled
  }
  #undef STAGE_TILE

  // ---- Butterfly min-reduce across the 16 lanes of each half (columns),
  // smaller centroid index wins ties (matches argmin's first-index rule).
  #pragma unroll
  for (int m = 1; m < 16; m <<= 1) {
    #pragma unroll
    for (int v = 0; v < 8; ++v) {
      float ov = __shfl_xor(minv[v], m, 32);
      int   oi = __shfl_xor(mini[v], m, 32);
      if (ov < minv[v] || (ov == minv[v] && oi < mini[v])) {
        minv[v] = ov;
        mini[v] = oi;
      }
    }
  }

  // Lane 0 holds rows m0+0..7, lane 16 holds rows m0+8..15.
  if (row == 0) {
    int rbase = m0 + halfId * 8;
    #pragma unroll
    for (int v = 0; v < 8; ++v) {
      float xn2 = lds_xn[wave * 16 + halfId * 8 + v];
      float sq  = xn2 + minv[v];
      mind[rbase + v]    = sqrtf(fmaxf(sq, 0.f));
      nearest[rbase + v] = mini[v];
    }
  }
}

// ---------------------------------------------------------------------------
// Kernel 3: per-batch threshold = mean + unbiased std of min distances.
// One block per batch.
// ---------------------------------------------------------------------------
__global__ __launch_bounds__(256) void asmk_stats(const float* __restrict__ mind,
                                                  float* __restrict__ thr) {
  int b = blockIdx.x;
  const float* md = mind + (size_t)b * NN;
  float s = 0.f, s2 = 0.f;
  for (int i = threadIdx.x; i < NN; i += 256) {
    float v = md[i];
    s += v;
    s2 += v * v;
  }
  #pragma unroll
  for (int m = 16; m >= 1; m >>= 1) {
    s += __shfl_xor(s, m, 32);
    s2 += __shfl_xor(s2, m, 32);
  }
  __shared__ float rs[8], rs2[8];
  int lane = threadIdx.x & 31, wv = threadIdx.x >> 5;
  if (lane == 0) { rs[wv] = s; rs2[wv] = s2; }
  __syncthreads();
  if (threadIdx.x == 0) {
    float S = 0.f, S2 = 0.f;
    for (int w = 0; w < 8; ++w) { S += rs[w]; S2 += rs2[w]; }
    float mean = S / (float)NN;
    float var  = (S2 - (float)NN * mean * mean) / (float)(NN - 1);  // ddof=1
    thr[b] = mean + sqrtf(fmaxf(var, 0.f));
  }
}

// ---------------------------------------------------------------------------
// Kernel 4a/4b: zero counts, then scatter-count masked assignments.
// ---------------------------------------------------------------------------
__global__ void zero_counts(float* __restrict__ counts) {
  counts[blockIdx.x * 256 + threadIdx.x] = 0.f;
}

__global__ void count_kernel(const float* __restrict__ mind,
                             const int* __restrict__ nearest,
                             const float* __restrict__ thr,
                             float* __restrict__ counts) {
  int i = blockIdx.x * 256 + threadIdx.x;  // 0 .. 65535
  int b = i >> 12;                         // / NN
  if (mind[i] < thr[b]) atomicAdd(&counts[b * KK + nearest[i]], 1.0f);
}

// ---------------------------------------------------------------------------
// Kernel 5: asmk = counts * weights; L2-normalize each batch row.
// ---------------------------------------------------------------------------
__global__ __launch_bounds__(256) void norm_kernel(const float* __restrict__ counts,
                                                   const float* __restrict__ w,
                                                   float* __restrict__ out) {
  int b = blockIdx.x;
  float s = 0.f;
  for (int c = threadIdx.x; c < KK; c += 256) {
    float v = counts[b * KK + c] * w[c];
    s += v * v;
  }
  #pragma unroll
  for (int m = 16; m >= 1; m >>= 1) s += __shfl_xor(s, m, 32);
  __shared__ float rs[8];
  __shared__ float nrm;
  int lane = threadIdx.x & 31, wv = threadIdx.x >> 5;
  if (lane == 0) rs[wv] = s;
  __syncthreads();
  if (threadIdx.x == 0) {
    float S = 0.f;
    for (int k = 0; k < 8; ++k) S += rs[k];
    nrm = fmaxf(sqrtf(S), 1e-12f);
  }
  __syncthreads();
  for (int c = threadIdx.x; c < KK; c += 256)
    out[b * KK + c] = counts[b * KK + c] * w[c] / nrm;
}

// ---------------------------------------------------------------------------
// Launch. Workspace layout (bytes):
//   ch      : 0        .. 262143   (1024*128 bf16 hi)
//   cl      : 262144   .. 524287   (1024*128 bf16 lo)
//   cnorm2  : 524288   .. 528383   (1024 f32)
//   mind    : 528384   .. 790527   (65536 f32)
//   nearest : 790528   .. 1052671  (65536 i32)
//   thr     : 1052672  .. 1052927  (16 f32, padded)
//   counts  : 1052928  .. 1118463  (16*1024 f32)
// ---------------------------------------------------------------------------
extern "C" void kernel_launch(void* const* d_in, const int* in_sizes, int n_in,
                              void* d_out, int out_size, void* d_ws, size_t ws_size,
                              hipStream_t stream) {
  const float* x   = (const float*)d_in[0];  // [16,4096,128]
  const float* cen = (const float*)d_in[1];  // [1024,128]
  const float* w   = (const float*)d_in[2];  // [1024]

  char* ws = (char*)d_ws;
  __bf16* ch      = (__bf16*)(ws + 0);
  __bf16* cl      = (__bf16*)(ws + 262144);
  float*  cnorm2  = (float*)(ws + 524288);
  float*  mind    = (float*)(ws + 528384);
  int*    nearest = (int*)(ws + 790528);
  float*  thr     = (float*)(ws + 1052672);
  float*  counts  = (float*)(ws + 1052928);
  float*  out     = (float*)d_out;  // [16,1024]

  prep_centroids<<<KK, 128, 0, stream>>>(cen, ch, cl, cnorm2);
  asmk_main<<<MTOT / (8 * 16), 256, 0, stream>>>(x, ch, cl, cnorm2, mind, nearest);
  asmk_stats<<<BB, 256, 0, stream>>>(mind, thr);
  zero_counts<<<(BB * KK) / 256, 256, 0, stream>>>(counts);
  count_kernel<<<MTOT / 256, 256, 0, stream>>>(mind, nearest, thr, counts);
  norm_kernel<<<BB, 256, 0, stream>>>(counts, w, out);
}
